// StackGCNs_88648124991109
// MI455X (gfx1250) — compile-verified, matrix operand order checked by reading, not run on hardware
//
#include <hip/hip_runtime.h>
#include <hip/hip_bf16.h>

typedef __attribute__((ext_vector_type(16))) _Float16 v16h;
typedef __attribute__((ext_vector_type(8)))  _Float16 v8h;
typedef __attribute__((ext_vector_type(8)))  float    v8f;

#define DIM 128

// ---------------------------------------------------------------------------
// Pack a 128x128 f32 row-major W into per-lane WMMA B-fragments (f16).
// Fragment (kt, nt) holds B[k = kt*32 .. +31][n = nt*16 .. +15].
// Per the CDNA5 ISA 16-bit operand layout:
//   lane l<16 : elems 0..7 -> K = kt*32 + (0..7),  elems 8..15 -> K = kt*32 + (16..23), N = l
//   lane l>=16: elems 0..7 -> K = kt*32 + (8..15), elems 8..15 -> K = kt*32 + (24..31), N = l-16
// Linear layout: Wf[((kt*8+nt)*32 + lane)*16 + i]  -> one contiguous 32B load/lane.
// ---------------------------------------------------------------------------
__global__ __launch_bounds__(256) void pack_wfrag(const float* __restrict__ W,
                                                  _Float16* __restrict__ Wf) {
    int t = blockIdx.x * blockDim.x + threadIdx.x;
    if (t >= 32 * 32 * 16) return;           // 32 frags * 32 lanes * 16 halves
    int i    = t & 15;
    int lane = (t >> 4) & 31;
    int frag = t >> 9;
    int kt = frag >> 3, nt = frag & 7;
    int koff;
    if (lane < 16) koff = (i < 8) ? i        : (16 + (i - 8));
    else           koff = (i < 8) ? (8 + i)  : (24 + (i - 8));
    int k = kt * 32 + koff;
    int n = nt * 16 + (lane & 15);
    Wf[t] = (_Float16)W[k * DIM + n];
}

// f32 -> f16 conversion, 4 elems per thread (n must be a multiple of 4; it is).
__global__ __launch_bounds__(256) void f32_to_f16_v4(const float* __restrict__ in,
                                                     _Float16* __restrict__ out, int n) {
    int i = (blockIdx.x * blockDim.x + threadIdx.x) * 4;
    if (i >= n) return;
    float4 v = *(const float4*)(in + i);
    v8h dummy; (void)dummy;
    out[i + 0] = (_Float16)v.x;
    out[i + 1] = (_Float16)v.y;
    out[i + 2] = (_Float16)v.z;
    out[i + 3] = (_Float16)v.w;
}

// deg[i] = 1.0 (self loop)
__global__ __launch_bounds__(256) void deg_init(float* __restrict__ deg, int N) {
    int i = blockIdx.x * blockDim.x + threadIdx.x;
    if (i < N) deg[i] = 1.0f;
}
// deg[dst] += 1 for each edge
__global__ __launch_bounds__(256) void deg_count(const int* __restrict__ ei,
                                                 float* __restrict__ deg, int E) {
    int e = blockIdx.x * blockDim.x + threadIdx.x;
    if (e < E) atomicAdd(&deg[ei[E + e]], 1.0f);
}
// in-place deg -> rsqrt(deg)
__global__ __launch_bounds__(256) void deg_rsqrt(float* __restrict__ deg, int N) {
    int i = blockIdx.x * blockDim.x + threadIdx.x;
    if (i < N) {
        float d = deg[i];
        deg[i] = (d > 0.0f) ? rsqrtf(d) : 0.0f;
    }
}

// out[i*128 + j] = b[j]
__global__ __launch_bounds__(256) void bias_init(float* __restrict__ out,
                                                 const float* __restrict__ b, int n) {
    int i = blockIdx.x * blockDim.x + threadIdx.x;
    if (i < n) out[i] = b[i & (DIM - 1)];
}

// ---------------------------------------------------------------------------
// C[M,128] = A16[M,128] @ W  using v_wmma_f32_16x16x32_f16.
// One wave per 16-row strip; 8 N-tiles x 4 K-tiles; B pre-packed fragments.
// ---------------------------------------------------------------------------
__global__ __launch_bounds__(128) void gemm16_wmma(const _Float16* __restrict__ A16,
                                                   const _Float16* __restrict__ Bfrag,
                                                   float* __restrict__ C, int M) {
    const int lane = threadIdx.x & 31;
    const int wave = blockIdx.x * (blockDim.x >> 5) + (threadIdx.x >> 5);
    const int row0 = wave << 4;
    if (row0 >= M) return;

    const int arow  = row0 + (lane & 15);
    const int kbase = (lane & 16) ? 8 : 0;

    // A fragments for all 4 K-tiles: two contiguous 16B loads each, per the
    // 16-bit A 16x32 VGPR layout.
    v16h afrag[4];
#pragma unroll
    for (int kt = 0; kt < 4; ++kt) {
        const _Float16* p = A16 + (size_t)arow * DIM + kt * 32 + kbase;
        v8h lo = *(const v8h*)p;
        v8h hi = *(const v8h*)(p + 16);
#pragma unroll
        for (int i = 0; i < 8; ++i) {
            afrag[kt][i]     = lo[i];
            afrag[kt][i + 8] = hi[i];
        }
    }

    // C/D layout: VGPR r, lanes 0..15 -> (M = r,   N = lane)
    //                     lanes 16..31 -> (M = 8+r, N = lane-16)
    const int cm  = row0 + ((lane >> 4) << 3);
    const int cn0 = lane & 15;

#pragma unroll
    for (int nt = 0; nt < 8; ++nt) {
        v8f acc = {};
#pragma unroll
        for (int kt = 0; kt < 4; ++kt) {
            v16h b = *(const v16h*)(Bfrag + (size_t)(((kt << 3) + nt) * 32 + lane) * 16);
            acc = __builtin_amdgcn_wmma_f32_16x16x32_f16(false, afrag[kt], false, b,
                                                         (short)0, acc, false, false);
        }
        float* cp = C + (size_t)cm * DIM + (nt << 4) + cn0;
#pragma unroll
        for (int r = 0; r < 8; ++r) cp[(size_t)r * DIM] = acc[r];
    }
}

// ---------------------------------------------------------------------------
// One wave per message (edge or self loop): out[dst] += hw[src] * dinv[src]*dinv[dst]
// 4 floats/lane: coalesced float4 gather + 4 global f32 atomic adds.
// ---------------------------------------------------------------------------
__global__ __launch_bounds__(256) void scatter_edges(const float* __restrict__ hw,
                                                     const int* __restrict__ ei,
                                                     const float* __restrict__ dinv,
                                                     float* __restrict__ out,
                                                     int E, int N) {
    int w    = (blockIdx.x * blockDim.x + threadIdx.x) >> 5;
    int lane = threadIdx.x & 31;
    if (w >= E + N) return;
    int s, d;
    if (w < E) { s = ei[w]; d = ei[E + w]; }
    else       { s = d = w - E; }
    float nrm = dinv[s] * dinv[d];
    float4 v = *(const float4*)(hw + (size_t)s * DIM + lane * 4);
    float* o = out + (size_t)d * DIM + lane * 4;
    atomicAdd(o + 0, v.x * nrm);
    atomicAdd(o + 1, v.y * nrm);
    atomicAdd(o + 2, v.z * nrm);
    atomicAdd(o + 3, v.w * nrm);
}

static inline size_t alignup(size_t x) { return (x + 255) & ~(size_t)255; }

extern "C" void kernel_launch(void* const* d_in, const int* in_sizes, int n_in,
                              void* d_out, int out_size, void* d_ws, size_t ws_size,
                              hipStream_t stream) {
    const float* x  = (const float*)d_in[0];
    const int*   ei = (const int*)d_in[1];
    const float* W1 = (const float*)d_in[2];
    const float* b1 = (const float*)d_in[3];
    const float* W2 = (const float*)d_in[4];
    const float* b2 = (const float*)d_in[5];
    float* out = (float*)d_out;

    const int N = in_sizes[0] / DIM;   // 50000
    const int E = in_sizes[1] / 2;     // 800000
    const size_t ND = (size_t)N * DIM;

    // Workspace carve-out
    char* p = (char*)d_ws;
    float*    dinv = (float*)p;    p += alignup((size_t)N * 4);
    _Float16* x16  = (_Float16*)p; p += alignup(ND * 2);
    _Float16* wf1  = (_Float16*)p; p += alignup(32 * 32 * 16 * 2);
    _Float16* wf2  = (_Float16*)p; p += alignup(32 * 32 * 16 * 2);
    float*    hw   = (float*)p;    p += alignup(ND * 4);   // GEMM output (both layers)
    float*    h1   = (float*)p;    p += alignup(ND * 4);   // layer-1 aggregation
    _Float16* h116 = (_Float16*)p; p += alignup(ND * 2);

    const int strips = N / 16;                 // 50000/16 = 3125 exactly
    const int gemm_blocks = (strips + 3) / 4;  // 4 waves / 128-thread block
    const int msgs = E + N;
    const int scat_blocks = (msgs * 32 + 255) / 256;
    const int nd_blocks   = (int)((ND + 255) / 256);
    const int cvt_blocks  = (int)((ND / 4 + 255) / 256);

    // --- one-time-per-call prep ---
    pack_wfrag<<<(32 * 32 * 16 + 255) / 256, 256, 0, stream>>>(W1, wf1);
    pack_wfrag<<<(32 * 32 * 16 + 255) / 256, 256, 0, stream>>>(W2, wf2);
    f32_to_f16_v4<<<cvt_blocks, 256, 0, stream>>>(x, x16, (int)ND);
    deg_init<<<(N + 255) / 256, 256, 0, stream>>>(dinv, N);
    deg_count<<<(E + 255) / 256, 256, 0, stream>>>(ei, dinv, E);
    deg_rsqrt<<<(N + 255) / 256, 256, 0, stream>>>(dinv, N);

    // --- layer 1: hw = x @ W1 ; h1 = scatter(hw) + b1 ---
    gemm16_wmma<<<gemm_blocks, 128, 0, stream>>>(x16, wf1, hw, N);
    bias_init<<<nd_blocks, 256, 0, stream>>>(h1, b1, (int)ND);
    scatter_edges<<<scat_blocks, 256, 0, stream>>>(hw, ei, dinv, h1, E, N);

    // --- layer 2: hw = h1 @ W2 ; out = scatter(hw) + b2 ---
    f32_to_f16_v4<<<cvt_blocks, 256, 0, stream>>>(h1, h116, (int)ND);
    gemm16_wmma<<<gemm_blocks, 128, 0, stream>>>(h116, wf2, hw, N);
    bias_init<<<nd_blocks, 256, 0, stream>>>(out, b2, (int)ND);
    scatter_edges<<<scat_blocks, 256, 0, stream>>>(hw, ei, dinv, out, E, N);
}
